// TransformerEncoder_1786706395840
// MI455X (gfx1250) — matrix operand order repaired and checked by reading, not compiled, not run on hardware
//
#include <hip/hip_runtime.h>
#include <hip/hip_bf16.h>
#include <math.h>

typedef __attribute__((ext_vector_type(16))) __bf16 v16bf;
typedef __attribute__((ext_vector_type(8)))  float  v8f;

#define D_MODEL 1024
#define N_HEAD  16
#define D_HEAD  64
#define BATCH   8
#define SEQ     512
#define ROWS    (BATCH * SEQ)          // 4096
#define INV_SCALE 0.125f               // 1/sqrt(64)

// bf16 fragment: 16 bf16 elements = 32 bytes = 2 x uint4
union Frag {
    v16bf v;
    uint4 q[2];
};

__device__ __forceinline__ unsigned short f2bf(float f) {
    unsigned int u = __float_as_uint(f);
    unsigned int r = u + 0x7FFFu + ((u >> 16) & 1u);
    return (unsigned short)(r >> 16);
}

// A-operand layout: K groups {khalf*8 .. +7} and {16+khalf*8 .. +7}
__device__ __forceinline__ void load_a(Frag& f, const unsigned short* p) {
    const uint4* q = reinterpret_cast<const uint4*>(p);
    f.q[0] = q[0];
    f.q[1] = q[2];          // +16 elements
}
// B-operand layout: K groups {khalf*8 .. +7} and {16+khalf*8 .. +7}
__device__ __forceinline__ void load_b(Frag& f, const unsigned short* p) {
    const uint4* q = reinterpret_cast<const uint4*>(p);
    f.q[0] = q[0];
    f.q[1] = q[2];          // +16 elements
}

#define WMMA_BF16(A, B, C) \
    __builtin_amdgcn_wmma_f32_16x16x32_bf16(false, (A), false, (B), (short)0, (C), false, false)

// ---------------------------------------------------------------------------
// f32 -> bf16 conversion (grid-stride)
// ---------------------------------------------------------------------------
__global__ void cvt_f32_bf16(const float* __restrict__ in,
                             unsigned short* __restrict__ out, int n) {
    for (int i = blockIdx.x * blockDim.x + threadIdx.x; i < n;
         i += gridDim.x * blockDim.x)
        out[i] = f2bf(in[i]);
}

// ---------------------------------------------------------------------------
// QKV GEMM: C[4096,4096] = xb[4096,1024] * Wqkv^T.
// 16x64 tile per wave, double-buffered fragments (always ~10 loads in flight).
// Results scattered into Q/K1/K2 [B,H,S,64] and Vt [B,H,64,S] (all bf16).
// ---------------------------------------------------------------------------
__global__ __launch_bounds__(256) void qkv_gemm(
    const unsigned short* __restrict__ xb,
    const unsigned short* __restrict__ wb,
    unsigned short* __restrict__ Q,
    unsigned short* __restrict__ K1,
    unsigned short* __restrict__ K2,
    unsigned short* __restrict__ Vt) {
    const int lane  = threadIdx.x & 31;
    const int wave  = threadIdx.x >> 5;
    const int tile  = blockIdx.x * 8 + wave;     // 16384 tiles (256 mt x 64 ntg)
    const int mt    = tile >> 6;
    const int ntg   = tile & 63;                 // group of 64 output columns
    const int ln    = lane & 15;
    const int khalf = lane >> 4;

    const unsigned short* ap = xb + (size_t)(mt * 16 + ln) * D_MODEL + khalf * 8;
    const unsigned short* bp[4];
#pragma unroll
    for (int t = 0; t < 4; ++t)
        bp[t] = wb + (size_t)(ntg * 64 + t * 16 + ln) * D_MODEL + khalf * 8;

    v8f zero = {};
    v8f acc[4] = {zero, zero, zero, zero};

    Frag a0, a1, b0[4], b1[4];
    load_a(a0, ap);
#pragma unroll
    for (int t = 0; t < 4; ++t) load_b(b0[t], bp[t]);

    for (int k = 0; k < D_MODEL - 64; k += 64) {
        load_a(a1, ap + k + 32);
#pragma unroll
        for (int t = 0; t < 4; ++t) load_b(b1[t], bp[t] + k + 32);
#pragma unroll
        for (int t = 0; t < 4; ++t) acc[t] = WMMA_BF16(a0.v, b0[t].v, acc[t]);

        load_a(a0, ap + k + 64);
#pragma unroll
        for (int t = 0; t < 4; ++t) load_b(b0[t], bp[t] + k + 64);
#pragma unroll
        for (int t = 0; t < 4; ++t) acc[t] = WMMA_BF16(a1.v, b1[t].v, acc[t]);
    }
    // epilogue: k = D_MODEL-64 (a0/b0 already loaded), then final 32
    load_a(a1, ap + D_MODEL - 32);
#pragma unroll
    for (int t = 0; t < 4; ++t) load_b(b1[t], bp[t] + D_MODEL - 32);
#pragma unroll
    for (int t = 0; t < 4; ++t) acc[t] = WMMA_BF16(a0.v, b0[t].v, acc[t]);
#pragma unroll
    for (int t = 0; t < 4; ++t) acc[t] = WMMA_BF16(a1.v, b1[t].v, acc[t]);

    // scatter: n = ntg*64 + t*16 + ln ; section = q/k1/k2/v ; head-major
#pragma unroll
    for (int t = 0; t < 4; ++t) {
        const int n       = ntg * 64 + t * 16 + ln;
        const int section = n >> 10;             // 0..3
        const int d       = n & 1023;
        const int h       = d >> 6;
        const int di      = d & 63;
#pragma unroll
        for (int i = 0; i < 8; ++i) {
            const int mrow = mt * 16 + i + 8 * khalf;
            const int bb   = mrow >> 9;
            const int s    = mrow & 511;
            const unsigned short val = f2bf(acc[t][i]);
            const size_t bh = (size_t)bb * N_HEAD + h;
            if (section == 0)      Q [(bh * SEQ + s) * D_HEAD + di] = val;
            else if (section == 1) K1[(bh * SEQ + s) * D_HEAD + di] = val;
            else if (section == 2) K2[(bh * SEQ + s) * D_HEAD + di] = val;
            else                   Vt[(bh * D_HEAD + di) * SEQ + s] = val;
        }
    }
}

// ---------------------------------------------------------------------------
// Attention: one wave per (b, h, 16-row q tile). Dual QK^T with speaker
// selection + distance bias + padding mask -> softmax -> PV.
// ---------------------------------------------------------------------------
__global__ __launch_bounds__(32) void attn_kernel(
    const unsigned short* __restrict__ Q,
    const unsigned short* __restrict__ K1,
    const unsigned short* __restrict__ K2,
    const unsigned short* __restrict__ Vt,
    const int* __restrict__ mask,
    const int* __restrict__ qmask,
    const float* __restrict__ shiftp,
    const float* __restrict__ biasp,
    unsigned short* __restrict__ ctx) {
    __shared__ float          sc[16][SEQ];   // 32 KB score strip
    __shared__ unsigned short pp[16][SEQ];   // 16 KB bf16 probs

    const int lane  = threadIdx.x;
    const int qt    = blockIdx.x & 31;
    const int h     = (blockIdx.x >> 5) & 15;
    const int b     = blockIdx.x >> 9;
    const int bh    = b * N_HEAD + h;
    const int ln    = lane & 15;
    const int khalf = lane >> 4;
    const int qbase = qt * 16;

    const float shift = shiftp[0];
    const float bias  = biasp[0];

    // Q fragments for dh halves 0..31 / 32..63 (held in VGPRs all kernel)
    Frag qa[2];
    const unsigned short* qptr = Q + ((size_t)bh * SEQ + qbase + ln) * D_HEAD;
    load_a(qa[0], qptr + khalf * 8);
    load_a(qa[1], qptr + 32 + khalf * 8);

    // qmask per accumulator row (rows i + 8*khalf)
    int qrow[8];
#pragma unroll
    for (int i = 0; i < 8; ++i)
        qrow[i] = qmask[b * SEQ + qbase + i + 8 * khalf];

    // ---- scores: 32 k-tiles of 16 keys -----------------------------------
    for (int kt = 0; kt < 32; ++kt) {
        const unsigned short* k1p = K1 + ((size_t)bh * SEQ + kt * 16 + ln) * D_HEAD + khalf * 8;
        const unsigned short* k2p = K2 + ((size_t)bh * SEQ + kt * 16 + ln) * D_HEAD + khalf * 8;
        Frag b1[2], b2[2];
        load_b(b1[0], k1p);
        load_b(b1[1], k1p + 32);
        load_b(b2[0], k2p);
        load_b(b2[1], k2p + 32);

        v8f a1 = {}, a2 = {};
        a1 = WMMA_BF16(qa[0].v, b1[0].v, a1);
        a1 = WMMA_BF16(qa[1].v, b1[1].v, a1);
        a2 = WMMA_BF16(qa[0].v, b2[0].v, a2);
        a2 = WMMA_BF16(qa[1].v, b2[1].v, a2);

        const int   kpos = kt * 16 + ln;
        const int   kq   = qmask[b * SEQ + kpos];
        const bool  kpad = mask[b * SEQ + kpos] != 0;
#pragma unroll
        for (int i = 0; i < 8; ++i) {
            const int m    = i + 8 * khalf;
            const int qpos = qbase + m;
            float v = (qrow[i] == kq) ? a1[i] : a2[i];
            v *= INV_SCALE;
            const float dd = (float)(qpos - kpos);
            v -= (shift * dd * dd + bias);
            if (!kpad) v = -INFINITY;
            sc[m][kpos] = v;
        }
    }
    __syncthreads();

    // ---- softmax (lanes 0..15, one row each) -----------------------------
    if (lane < 16) {
        float mx = -INFINITY;
        for (int j = 0; j < SEQ; ++j) mx = fmaxf(mx, sc[lane][j]);
        float sum = 0.f;
        for (int j = 0; j < SEQ; ++j) {
            const float e = __expf(sc[lane][j] - mx);
            sc[lane][j] = e;
            sum += e;
        }
        const float inv = 1.0f / sum;
        for (int j = 0; j < SEQ; ++j) pp[lane][j] = f2bf(sc[lane][j] * inv);
    }
    __syncthreads();

    // ---- PV: out[16 x 64] = P[16 x 512] * V[512 x 64] --------------------
    v8f zero = {};
    v8f o[4] = {zero, zero, zero, zero};
    for (int k = 0; k < SEQ; k += 32) {
        Frag pa;
        load_a(pa, &pp[ln][0] + k + khalf * 8);
        Frag vb[4];
#pragma unroll
        for (int t = 0; t < 4; ++t)
            load_b(vb[t], Vt + ((size_t)bh * D_HEAD + t * 16 + ln) * SEQ + k + khalf * 8);
#pragma unroll
        for (int t = 0; t < 4; ++t)
            o[t] = WMMA_BF16(pa.v, vb[t].v, o[t]);
    }

    // ---- store context in [4096, 1024] layout (s-major, h*64+d cols) -----
#pragma unroll
    for (int t = 0; t < 4; ++t)
#pragma unroll
        for (int i = 0; i < 8; ++i) {
            const int m = i + 8 * khalf;
            const size_t row = (size_t)b * SEQ + qbase + m;
            ctx[row * D_MODEL + h * D_HEAD + t * 16 + ln] = f2bf(o[t][i]);
        }
}

// ---------------------------------------------------------------------------
// Output projection: out[4096,1024] = ctx * fc_w^T + fc_b (f32 out).
// 16x64 tile per wave, double-buffered fragments.
// ---------------------------------------------------------------------------
__global__ __launch_bounds__(256) void out_gemm(
    const unsigned short* __restrict__ ctx,
    const unsigned short* __restrict__ wb,
    const float* __restrict__ fcb,
    float* __restrict__ out) {
    const int lane  = threadIdx.x & 31;
    const int wave  = threadIdx.x >> 5;
    const int tile  = blockIdx.x * 8 + wave;     // 4096 tiles (256 mt x 16 ntg)
    const int mt    = tile >> 4;
    const int ntg   = tile & 15;
    const int ln    = lane & 15;
    const int khalf = lane >> 4;

    const unsigned short* ap = ctx + (size_t)(mt * 16 + ln) * D_MODEL + khalf * 8;
    const unsigned short* bp[4];
#pragma unroll
    for (int t = 0; t < 4; ++t)
        bp[t] = wb + (size_t)(ntg * 64 + t * 16 + ln) * D_MODEL + khalf * 8;

    v8f zero = {};
    v8f acc[4] = {zero, zero, zero, zero};

    Frag a0, a1, b0[4], b1[4];
    load_a(a0, ap);
#pragma unroll
    for (int t = 0; t < 4; ++t) load_b(b0[t], bp[t]);

    for (int k = 0; k < D_MODEL - 64; k += 64) {
        load_a(a1, ap + k + 32);
#pragma unroll
        for (int t = 0; t < 4; ++t) load_b(b1[t], bp[t] + k + 32);
#pragma unroll
        for (int t = 0; t < 4; ++t) acc[t] = WMMA_BF16(a0.v, b0[t].v, acc[t]);

        load_a(a0, ap + k + 64);
#pragma unroll
        for (int t = 0; t < 4; ++t) load_b(b0[t], bp[t] + k + 64);
#pragma unroll
        for (int t = 0; t < 4; ++t) acc[t] = WMMA_BF16(a1.v, b1[t].v, acc[t]);
    }
    load_a(a1, ap + D_MODEL - 32);
#pragma unroll
    for (int t = 0; t < 4; ++t) load_b(b1[t], bp[t] + D_MODEL - 32);
#pragma unroll
    for (int t = 0; t < 4; ++t) acc[t] = WMMA_BF16(a0.v, b0[t].v, acc[t]);
#pragma unroll
    for (int t = 0; t < 4; ++t) acc[t] = WMMA_BF16(a1.v, b1[t].v, acc[t]);

#pragma unroll
    for (int t = 0; t < 4; ++t) {
        const int   n  = ntg * 64 + t * 16 + ln;
        const float bn = fcb[n];
#pragma unroll
        for (int i = 0; i < 8; ++i) {
            const int mrow = mt * 16 + i + 8 * khalf;
            out[(size_t)mrow * D_MODEL + n] = acc[t][i] + bn;
        }
    }
}

// ---------------------------------------------------------------------------
extern "C" void kernel_launch(void* const* d_in, const int* in_sizes, int n_in,
                              void* d_out, int out_size, void* d_ws, size_t ws_size,
                              hipStream_t stream) {
    const float* x     = (const float*)d_in[0];
    const int*   mask  = (const int*)d_in[1];
    const int*   qmask = (const int*)d_in[2];
    const float* Wqkv  = (const float*)d_in[3];
    const float* fc_w  = (const float*)d_in[4];
    const float* fc_b  = (const float*)d_in[5];
    const float* shift = (const float*)d_in[6];
    const float* bias  = (const float*)d_in[7];

    unsigned short* xb    = (unsigned short*)d_ws;
    unsigned short* wqkvb = xb    + (size_t)ROWS * D_MODEL;          // 4M
    unsigned short* fcwb  = wqkvb + (size_t)4 * D_MODEL * D_MODEL;   // 4M
    unsigned short* Qb    = fcwb  + (size_t)D_MODEL * D_MODEL;       // 1M
    unsigned short* K1b   = Qb    + (size_t)ROWS * D_MODEL;
    unsigned short* K2b   = K1b   + (size_t)ROWS * D_MODEL;
    unsigned short* Vtb   = K2b   + (size_t)ROWS * D_MODEL;
    unsigned short* ctx   = Vtb   + (size_t)ROWS * D_MODEL;

    cvt_f32_bf16<<<1024, 256, 0, stream>>>(x, xb, ROWS * D_MODEL);
    cvt_f32_bf16<<<1024, 256, 0, stream>>>(Wqkv, wqkvb, 4 * D_MODEL * D_MODEL);
    cvt_f32_bf16<<<1024, 256, 0, stream>>>(fc_w, fcwb, D_MODEL * D_MODEL);

    qkv_gemm<<<2048, 256, 0, stream>>>(xb, wqkvb, Qb, K1b, K2b, Vtb);
    attn_kernel<<<BATCH * N_HEAD * (SEQ / 16), 32, 0, stream>>>(
        Qb, K1b, K2b, Vtb, mask, qmask, shift, bias, ctx);
    out_gemm<<<512, 256, 0, stream>>>(ctx, fcwb, fc_b, (float*)d_out);
}
